// ScalarDotProductPolicyNetworkRevised_78649441124616
// MI455X (gfx1250) — matrix operand order repaired and checked by reading, not compiled, not run on hardware
//
#include <hip/hip_runtime.h>
#include <hip/hip_bf16.h>

typedef __bf16 bf16_t;
typedef __attribute__((ext_vector_type(16))) __bf16 v16bf;
typedef __attribute__((ext_vector_type(8)))  float  v8f;

namespace {
constexpr int kAgents = 32;    // N
constexpr int kDin    = 128;   // D_IN == D_PROC == D_FIN
constexpr int kHid    = 256;
constexpr int kAct    = 10;
constexpr int kActPad = 16;

constexpr int kWavesPerBlock = 4;

// per-wave LDS slab layout (bytes)
constexpr int kOffX   = 0;      // 32x128 bf16: states / wav / node_features
constexpr int kOffQ   = 8192;   // 32x128 bf16 (also low half of 32x256 H)
constexpr int kOffK   = 16384;  // 32x128 bf16 (also high half of H)
constexpr int kOffV   = 24576;  // 128x32 bf16 (V stored TRANSPOSED, ld = 32)
constexpr int kOffS   = 32768;  // 32x32 f32 scores/probs; later 32x16 f32 logits
constexpr int kPerWave = 36864;

// bf16 weight offsets inside d_ws (elements)
constexpr int kWk1 = 0;
constexpr int kWq1 = 16384;
constexpr int kWv1 = 32768;
constexpr int kWk2 = 49152;
constexpr int kWq2 = 65536;
constexpr int kWv2 = 81920;
constexpr int kWp1 = 98304;   // 256x128
constexpr int kWp2 = 131072;  // 16x256 (rows 10..15 zero)
} // namespace

// ---------------- WMMA fragment helpers (wave32, 16x16x32 bf16) ----------------

__device__ inline v8f wmma_bf16(v16bf a, v16bf b, v8f c) {
  return __builtin_amdgcn_wmma_f32_16x16x32_bf16(false, a, false, b, (short)0, c,
                                                 false, false);
}

// A fragment from row-major bf16 (leading dim ld); p -> tile origin (row0,k0).
// 16-bit A 16x32: lanes 0-15 row=lane, K = 0..7,16..23; lanes 16-31 K = 8..15,24..31.
__device__ inline v16bf ldA_bf16(const bf16_t* p, int ld, int lane) {
  const int row = lane & 15;
  const int kb  = (lane & 16) ? 8 : 0;
  const bf16_t* r = p + row * ld + kb;
  v16bf a;
#pragma unroll
  for (int i = 0; i < 8; ++i) { a[i] = r[i]; a[i + 8] = r[16 + i]; }
  return a;
}

// Same A layout, fp32 source (softmax probabilities), convert on load.
__device__ inline v16bf ldA_f32(const float* p, int ld, int lane) {
  const int row = lane & 15;
  const int kb  = (lane & 16) ? 8 : 0;
  const float* r = p + row * ld + kb;
  v16bf a;
#pragma unroll
  for (int i = 0; i < 8; ++i) {
    a[i]     = (bf16_t)r[i];
    a[i + 8] = (bf16_t)r[16 + i];
  }
  return a;
}

// B fragment where B = W^T, W row-major (Nout x K): two runs of 8 bf16 per lane, 16 apart.
// 16-bit B 32x16: lanes 0-15 n=lane K=0..7,16..23; lanes 16-31 n=lane-16 K=8..15,24..31.
__device__ inline v16bf ldB_wT(const bf16_t* w, int ld, int lane) {
  const int n  = lane & 15;
  const int kb = (lane & 16) ? 8 : 0;
  const bf16_t* r = w + n * ld + kb;
  v16bf b;
#pragma unroll
  for (int i = 0; i < 8; ++i) { b[i] = r[i]; b[i + 8] = r[16 + i]; }
  return b;
}

// D store, row-major: lane&15 = N column, lane&16 selects M rows 0-7 vs 8-15.
__device__ inline void stD_bf16(bf16_t* p, int ld, int lane, v8f d) {
  const int n  = lane & 15;
  const int mb = (lane & 16) ? 8 : 0;
#pragma unroll
  for (int j = 0; j < 8; ++j) p[(mb + j) * ld + n] = (bf16_t)d[j];
}

// D store, TRANSPOSED: writes D[m][n] to p[n*ldT + m] -> 8 consecutive bf16/lane
// (vectorizes into one 16B ds_store after v_cvt_pk_bf16_f32 packing).
__device__ inline void stD_bf16_T(bf16_t* p, int ldT, int lane, v8f d) {
  const int n  = lane & 15;
  const int mb = (lane & 16) ? 8 : 0;
  bf16_t* r = p + n * ldT + mb;
#pragma unroll
  for (int j = 0; j < 8; ++j) r[j] = (bf16_t)d[j];
}

__device__ inline void stD_f32(float* p, int ld, int lane, v8f d, float scale) {
  const int n  = lane & 15;
  const int mb = (lane & 16) ? 8 : 0;
#pragma unroll
  for (int j = 0; j < 8; ++j) p[(mb + j) * ld + n] = d[j] * scale;
}

// Out(32 x NOUT) = X(32 x K) @ W^T.
// A-fragments for BOTH 16-row tiles are hoisted into registers; each B fragment
// is loaded once and feeds two WMMAs. Optional fused leaky-ReLU; optional
// transposed output (Out laid out [NOUT][32], ld = 32).
template <int K, int NOUT, bool LRELU, bool TSTORE>
__device__ inline void gemm_xwT(const bf16_t* X, const bf16_t* W, bf16_t* Out,
                                int lane) {
  constexpr int KS = K / 32;
  v16bf a0[KS], a1[KS];
#pragma unroll
  for (int ks = 0; ks < KS; ++ks) {
    a0[ks] = ldA_bf16(X + ks * 32, K, lane);
    a1[ks] = ldA_bf16(X + 16 * K + ks * 32, K, lane);
  }
  for (int tn = 0; tn < NOUT / 16; ++tn) {
    v8f acc0 = {0.f, 0.f, 0.f, 0.f, 0.f, 0.f, 0.f, 0.f};
    v8f acc1 = {0.f, 0.f, 0.f, 0.f, 0.f, 0.f, 0.f, 0.f};
#pragma unroll
    for (int ks = 0; ks < KS; ++ks) {
      v16bf b = ldB_wT(W + tn * 16 * K + ks * 32, K, lane);
      acc0 = wmma_bf16(a0[ks], b, acc0);
      acc1 = wmma_bf16(a1[ks], b, acc1);
    }
    if (LRELU) {
#pragma unroll
      for (int j = 0; j < 8; ++j) {
        acc0[j] = acc0[j] >= 0.f ? acc0[j] : 0.01f * acc0[j];
        acc1[j] = acc1[j] >= 0.f ? acc1[j] : 0.01f * acc1[j];
      }
    }
    if (TSTORE) {
      stD_bf16_T(Out + tn * 16 * kAgents + 0, kAgents, lane, acc0);
      stD_bf16_T(Out + tn * 16 * kAgents + 16, kAgents, lane, acc1);
    } else {
      stD_bf16(Out + tn * 16, NOUT, lane, acc0);
      stD_bf16(Out + 16 * NOUT + tn * 16, NOUT, lane, acc1);
    }
  }
}

// S(32x32, f32) = scale * Q(32x128) @ K1^T  (K1 row-major => W^T-style B loads)
__device__ inline void attn_scores(const bf16_t* Q, const bf16_t* Kmat, float* S,
                                   float scale, int lane) {
  v16bf a0[4], a1[4];
#pragma unroll
  for (int ks = 0; ks < 4; ++ks) {
    a0[ks] = ldA_bf16(Q + ks * 32, kDin, lane);
    a1[ks] = ldA_bf16(Q + 16 * kDin + ks * 32, kDin, lane);
  }
  for (int tn = 0; tn < 2; ++tn) {
    v8f acc0 = {0.f, 0.f, 0.f, 0.f, 0.f, 0.f, 0.f, 0.f};
    v8f acc1 = {0.f, 0.f, 0.f, 0.f, 0.f, 0.f, 0.f, 0.f};
#pragma unroll
    for (int ks = 0; ks < 4; ++ks) {
      v16bf b = ldB_wT(Kmat + tn * 16 * kDin + ks * 32, kDin, lane);
      acc0 = wmma_bf16(a0[ks], b, acc0);
      acc1 = wmma_bf16(a1[ks], b, acc1);
    }
    stD_f32(S + tn * 16, kAgents, lane, acc0, scale);
    stD_f32(S + 16 * kAgents + tn * 16, kAgents, lane, acc1, scale);
  }
}

// Row softmax of S (32x32, one row per lane) in LDS, then coalesced float4
// streaming of the probability tile to global (attention-weight output).
__device__ inline void softmax32(float* S, float* __restrict__ gout, int lane) {
  float* r = S + lane * kAgents;
  float m = r[0];
#pragma unroll
  for (int i = 1; i < kAgents; ++i) m = fmaxf(m, r[i]);
  float sum = 0.f;
#pragma unroll
  for (int i = 0; i < kAgents; ++i) {
    float e = __expf(r[i] - m);
    r[i] = e;
    sum += e;
  }
  const float inv = 1.f / sum;
#pragma unroll
  for (int i = 0; i < kAgents; ++i) r[i] *= inv;

  const float4* s4 = (const float4*)S;      // 256 float4 = 32x32 fp32
  float4* g4 = (float4*)gout;
#pragma unroll
  for (int i = 0; i < 8; ++i) g4[i * 32 + lane] = s4[i * 32 + lane];
}

// Out(32x128) = P(32x32 f32) @ V ; V held TRANSPOSED (Vt[e][m], ld=32) so the
// B fragment is read from one Vt row per lane (two 8-element runs).
__device__ inline void pv(const float* S, const bf16_t* Vt, bf16_t* Out, int lane) {
  const v16bf a0 = ldA_f32(S, kAgents, lane);
  const v16bf a1 = ldA_f32(S + 16 * kAgents, kAgents, lane);
  for (int tn = 0; tn < 8; ++tn) {
    v16bf b = ldB_wT(Vt + tn * 16 * kAgents, kAgents, lane);
    v8f z = {0.f, 0.f, 0.f, 0.f, 0.f, 0.f, 0.f, 0.f};
    v8f d0 = wmma_bf16(a0, b, z);
    v8f d1 = wmma_bf16(a1, b, z);
    stD_bf16(Out + tn * 16, kDin, lane, d0);
    stD_bf16(Out + 16 * kDin + tn * 16, kDin, lane, d1);
  }
}

// ---------------- main fused kernel: one wave == one batch element ----------------

__global__ void __launch_bounds__(kWavesPerBlock * 32)
policy_attn_kernel(const float* __restrict__ states, const bf16_t* __restrict__ ws,
                   float* __restrict__ out_policy, float* __restrict__ out_w1,
                   float* __restrict__ out_w2, int nbatch) {
  extern __shared__ __align__(16) char smem[];
  const int wave = threadIdx.x >> 5;
  const int lane = threadIdx.x & 31;
  const int batch = blockIdx.x * kWavesPerBlock + wave;
  if (batch >= nbatch) return;  // wave-uniform: EXEC stays all-ones for WMMA

  char* base = smem + wave * kPerWave;
  bf16_t* X  = (bf16_t*)(base + kOffX);
  bf16_t* Q  = (bf16_t*)(base + kOffQ);
  bf16_t* Kb = (bf16_t*)(base + kOffK);
  bf16_t* Vt = (bf16_t*)(base + kOffV);  // transposed V: [128][32]
  float*  S  = (float*)(base + kOffS);
  bf16_t* H  = (bf16_t*)(base + kOffQ);  // 32x256, overlays Q+Kb

  const bf16_t* Wk1 = ws + kWk1;
  const bf16_t* Wq1 = ws + kWq1;
  const bf16_t* Wv1 = ws + kWv1;
  const bf16_t* Wk2 = ws + kWk2;
  const bf16_t* Wq2 = ws + kWq2;
  const bf16_t* Wv2 = ws + kWv2;
  const bf16_t* Wp1 = ws + kWp1;
  const bf16_t* Wp2 = ws + kWp2;

  // warm the (L2-resident) bf16 weight lines
  __builtin_prefetch(ws + (size_t)lane * 256, 0, 1);

  // stage states tile into LDS as bf16 (coalesced)
  const float* src = states + (size_t)batch * (kAgents * kDin);
#pragma unroll 4
  for (int i = lane; i < kAgents * kDin; i += 32) X[i] = (bf16_t)src[i];

  const float inv_scale = 0.08838834764831845f;  // 1/sqrt(128) for BOTH blocks

  // ---- attention block 1 ----
  gemm_xwT<kDin, kDin, false, false>(X, Wq1, Q, lane);
  gemm_xwT<kDin, kDin, false, false>(X, Wk1, Kb, lane);
  gemm_xwT<kDin, kDin, false, true >(X, Wv1, Vt, lane);
  attn_scores(Q, Kb, S, inv_scale, lane);
  softmax32(S, out_w1 + (size_t)batch * kAgents * kAgents, lane);
  pv(S, Vt, X, lane);  // X := wav

  // ---- attention block 2 ----
  gemm_xwT<kDin, kDin, false, false>(X, Wq2, Q, lane);
  gemm_xwT<kDin, kDin, false, false>(X, Wk2, Kb, lane);
  gemm_xwT<kDin, kDin, false, true >(X, Wv2, Vt, lane);
  attn_scores(Q, Kb, S, inv_scale, lane);
  softmax32(S, out_w2 + (size_t)batch * kAgents * kAgents, lane);
  pv(S, Vt, X, lane);  // X := node_features

  // ---- policy MLP head ----
  gemm_xwT<kDin, kHid, true, false>(X, Wp1, H, lane);  // fused leaky_relu

  float* L = S;  // 32 x 16 logits (padded), reuse score buffer
  for (int tm = 0; tm < 2; ++tm) {
    v8f acc = {0.f, 0.f, 0.f, 0.f, 0.f, 0.f, 0.f, 0.f};
#pragma unroll
    for (int ks = 0; ks < kHid / 32; ++ks) {
      v16bf a = ldA_bf16(H + tm * 16 * kHid + ks * 32, kHid, lane);
      v16bf b = ldB_wT(Wp2 + ks * 32, kHid, lane);
      acc = wmma_bf16(a, b, acc);
    }
    stD_f32(L + tm * 16 * kActPad, kActPad, lane, acc, 1.f);
  }

  // softmax over the 10 real actions (row = lane), normalize in LDS,
  // then coalesced store of the 32x10 policy rows.
  {
    float* r = L + lane * kActPad;
    float m = r[0];
#pragma unroll
    for (int i = 1; i < kAct; ++i) m = fmaxf(m, r[i]);
    float sum = 0.f;
#pragma unroll
    for (int i = 0; i < kAct; ++i) {
      float e = __expf(r[i] - m);
      r[i] = e;
      sum += e;
    }
    const float inv = 1.f / sum;
#pragma unroll
    for (int i = 0; i < kAct; ++i) r[i] *= inv;

    float* gp = out_policy + (size_t)batch * kAgents * kAct;
#pragma unroll
    for (int t = 0; t < kAct; ++t) {  // 320 floats, lane-striped
      int idx = t * 32 + lane;
      int row = idx / kAct;
      int col = idx - row * kAct;
      gp[idx] = L[row * kActPad + col];
    }
  }
}

// ---------------- weight prep: fp32 -> bf16 (Wp2 padded to 16x256) ----------------

__global__ void cvt_bf16_kernel(const float* __restrict__ src,
                                bf16_t* __restrict__ dst, int n) {
  int i = blockIdx.x * blockDim.x + threadIdx.x;
  if (i < n) dst[i] = (bf16_t)src[i];
}

__global__ void wp2_pad_kernel(const float* __restrict__ src,
                               bf16_t* __restrict__ dst) {
  int i = blockIdx.x * blockDim.x + threadIdx.x;  // 16*256 = 4096 threads
  if (i < kActPad * kHid) {
    int r = i >> 8, c = i & 255;
    dst[i] = (r < kAct) ? (bf16_t)src[r * kHid + c] : (bf16_t)0.0f;
  }
}

// ---------------- host entry ----------------

extern "C" void kernel_launch(void* const* d_in, const int* in_sizes, int n_in,
                              void* d_out, int out_size, void* d_ws, size_t ws_size,
                              hipStream_t stream) {
  (void)n_in; (void)out_size; (void)ws_size;

  const float* states = (const float*)d_in[0];
  const int nbatch = in_sizes[0] / (kAgents * kDin);  // 8192

  bf16_t* ws = (bf16_t*)d_ws;

  const int woff[7] = {kWk1, kWq1, kWv1, kWk2, kWq2, kWv2, kWp1};
  const int wlen[7] = {16384, 16384, 16384, 16384, 16384, 16384, 32768};
  for (int w = 0; w < 7; ++w) {
    cvt_bf16_kernel<<<(wlen[w] + 255) / 256, 256, 0, stream>>>(
        (const float*)d_in[1 + w], ws + woff[w], wlen[w]);
  }
  wp2_pad_kernel<<<(kActPad * kHid + 255) / 256, 256, 0, stream>>>(
      (const float*)d_in[8], ws + kWp2);

  // Output layout: policy | weight_1 | weight
  float* out_policy = (float*)d_out;
  float* out_w1 = out_policy + (size_t)nbatch * kAgents * kAct;
  float* out_w2 = out_w1 + (size_t)nbatch * kAgents * kAgents;

  dim3 grid((nbatch + kWavesPerBlock - 1) / kWavesPerBlock);
  dim3 block(kWavesPerBlock * 32);
  size_t lds_bytes = (size_t)kWavesPerBlock * kPerWave;  // 144 KB / block
  policy_attn_kernel<<<grid, block, lds_bytes, stream>>>(
      states, ws, out_policy, out_w1, out_w2, nbatch);
}
